// EmbeddingMixer_28647431865058
// MI455X (gfx1250) — matrix operand order repaired and checked by reading, not compiled, not run on hardware
//
#include <hip/hip_runtime.h>
#include <hip/hip_bf16.h>
#include <math.h>

// ---------------------------------------------------------------------------
// KAN 2-layer forward for MI455X (gfx1250), wave32 WMMA f32_16x16x32_f16.
//   N=8192 tokens, D=256, H=1024, spline order 3, grid 3 -> 6 basis fns.
//   Each layer == GEMM over augmented K = feat*7 (silu + 6 cardinal B-splines).
//   Block = 8 waves computing a 128x64 tile; B panel staged in LDS via
//   double-buffered GLOBAL_LOAD_ASYNC_TO_LDS_B128 (ASYNCcnt), A direct global.
// ---------------------------------------------------------------------------

typedef _Float16 v16h __attribute__((ext_vector_type(16)));
typedef _Float16 v8h  __attribute__((ext_vector_type(8)));
typedef float    v8f  __attribute__((ext_vector_type(8)));

#define DIN    256
#define HID    1024
#define NTOK   8192
#define CHUNK  2048
#define NCHUNK (NTOK / CHUNK)
#define K1     (DIN * 7)   // 1792 = 56*32
#define K2     (HID * 7)   // 7168 = 224*32

#define BROWS   32                  // K rows staged per step
#define BCOLS   64                  // N columns per block tile
#define BPAD    8                   // pad halves -> 144B row stride (bank spread)
#define BSTRIDE (BCOLS + BPAD)      // 72 halves
#define BUFH    (BROWS * BSTRIDE)   // halves per LDS buffer

// ---------------------------------------------------------------------------
// scalar helpers
// ---------------------------------------------------------------------------
__device__ __forceinline__ float kan_silu(float v) {
  return v / (1.0f + __expf(-v));
}
__device__ __forceinline__ float kan_gelu(float v) {
  return 0.5f * v * (1.0f + erff(v * 0.70710678118654752f));
}

// Uniform cubic B-spline basis on knots t_j = j*(2/3) - 1, j = -3..6.
// Zero outside [-3,3); otherwise 4 cardinal segment weights scattered into
// the 6 retained basis columns (matches Cox-de Boor on the uniform grid).
__device__ __forceinline__ void kan_basis(float g, float out[6]) {
#pragma unroll
  for (int k = 0; k < 6; ++k) out[k] = 0.0f;
  float t = (g + 3.0f) * 1.5f;           // (g - (-3)) / h, h = 2/3
  int a = (int)floorf(t);
  if (a < 0 || a > 8) return;
  float u  = t - (float)a;
  float um = 1.0f - u;
  float u2 = u * u, u3 = u2 * u;
  float w[4];
  w[0] = um * um * um * (1.0f / 6.0f);
  w[1] = (3.0f * u3 - 6.0f * u2 + 4.0f) * (1.0f / 6.0f);
  w[2] = (-3.0f * u3 + 3.0f * u2 + 3.0f * u + 1.0f) * (1.0f / 6.0f);
  w[3] = u3 * (1.0f / 6.0f);
#pragma unroll
  for (int s = 0; s < 4; ++s) {
    int idx = a - 3 + s;
    if (idx >= 0 && idx < 6) out[idx] = w[s];
  }
}

// ---------------------------------------------------------------------------
// gfx1250 async copy: 16B global -> LDS, tracked by ASYNCcnt.
// Inline asm per CDNA5 ISA (portable across toolchains).
// ---------------------------------------------------------------------------
__device__ __forceinline__ void kan_async_b128(const void* gsrc, void* ldst) {
  unsigned lds = (unsigned)(uintptr_t)ldst;                // low 32 bits = LDS addr
  unsigned long long ga = (unsigned long long)(uintptr_t)gsrc;
  asm volatile("global_load_async_to_lds_b128 %0, %1, off"
               :: "v"(lds), "v"(ga) : "memory");
}
__device__ __forceinline__ void kan_wait_async() {
  asm volatile("s_wait_asynccnt 0x0" ::: "memory");
}

// ---------------------------------------------------------------------------
// Weight packing: augmented K-major f16 weight matrices.
//   W[(i*7+0)*ldo + o] = base_w[o,i]
//   W[(i*7+1+j)*ldo + o] = spline_w[o,i,j] * scaler[o,i]
// ---------------------------------------------------------------------------
__global__ void kan_pack_w1(const float* __restrict__ bw,
                            const float* __restrict__ sw,
                            const float* __restrict__ sc,
                            _Float16* __restrict__ W) {
  int idx = blockIdx.x * blockDim.x + threadIdx.x;   // o*DIN + i, o<HID
  if (idx >= HID * DIN) return;
  int o = idx / DIN, i = idx % DIN;
  float s = sc[idx];
  W[(size_t)(i * 7) * HID + o] = (_Float16)bw[idx];
#pragma unroll
  for (int j = 0; j < 6; ++j)
    W[(size_t)(i * 7 + 1 + j) * HID + o] = (_Float16)(sw[(size_t)idx * 6 + j] * s);
}

__global__ void kan_pack_w2(const float* __restrict__ bw,
                            const float* __restrict__ sw,
                            const float* __restrict__ sc,
                            _Float16* __restrict__ W) {
  int idx = blockIdx.x * blockDim.x + threadIdx.x;   // o*HID + i, o<DIN
  if (idx >= DIN * HID) return;
  int o = idx / HID, i = idx % HID;
  float s = sc[idx];
  W[(size_t)(i * 7) * DIN + o] = (_Float16)bw[idx];
#pragma unroll
  for (int j = 0; j < 6; ++j)
    W[(size_t)(i * 7 + 1 + j) * DIN + o] = (_Float16)(sw[(size_t)idx * 6 + j] * s);
}

// ---------------------------------------------------------------------------
// Augmented activation build for layer 1 (one chunk of tokens).
// ---------------------------------------------------------------------------
__global__ void kan_build_a1(const float* __restrict__ x,
                             _Float16* __restrict__ A1, int tok0) {
  int idx = blockIdx.x * blockDim.x + threadIdx.x;   // n*DIN + i, n<CHUNK
  if (idx >= CHUNK * DIN) return;
  int n = idx / DIN, i = idx % DIN;
  float xv = x[(size_t)(tok0 + n) * DIN + i];
  _Float16* dst = A1 + (size_t)n * K1 + (size_t)i * 7;
  dst[0] = (_Float16)kan_silu(xv);
  float bas[6];
  kan_basis(xv, bas);
#pragma unroll
  for (int j = 0; j < 6; ++j) dst[1 + j] = (_Float16)bas[j];
}

// ---------------------------------------------------------------------------
// Block-level WMMA GEMM core: 8 waves x (16x64) = 128x64 output tile.
// A: MxK row-major f16 (direct global loads, ISA A-matrix 16x32 layout:
//    lane = M row; per-lane halves at K = kt+8*half and kt+16+8*half).
// B: KxN row-major f16, staged 32x64 per step into LDS by async DMA,
//    double-buffered; fragment read = lane-indexed N column, 16 K rows (table order).
// ---------------------------------------------------------------------------
template <int KDIM, int LDB>
__device__ __forceinline__ void kan_wmma_block(const _Float16* __restrict__ A,
                                               const _Float16* __restrict__ B,
                                               _Float16* smem,       // [2][BUFH]
                                               int mBaseWave, int nBaseBlk,
                                               v8f acc[4]) {
  const int tid   = threadIdx.x;
  const int lane  = tid & 31;
  const int halfk = (lane >> 4) * 8;
  const _Float16* arow = A + (size_t)(mBaseWave + (lane & 15)) * KDIM;

  // staging map: thread -> one 16B slice of the 32x64 panel
  const int srow = tid >> 3;                       // 0..31 (K row)
  const int sseg = (tid & 7) * 8;                  // halves within row

  // prologue: kick off first panel
  kan_async_b128(B + (size_t)srow * LDB + nBaseBlk + sseg,
                 smem + srow * BSTRIDE + sseg);

  int cur = 0;
  for (int kt = 0; kt < KDIM; kt += 32) {
    kan_wait_async();            // our slice of the current panel landed
    __syncthreads();             // all slices landed

    if (kt + 32 < KDIM) {        // stream next panel into the other buffer
      kan_async_b128(B + (size_t)(kt + 32 + srow) * LDB + nBaseBlk + sseg,
                     smem + (cur ^ 1) * BUFH + srow * BSTRIDE + sseg);
    }

    // A fragment (global)
    v8h alo = *(const v8h*)(arow + kt + halfk);
    v8h ahi = *(const v8h*)(arow + kt + 16 + halfk);
    v16h a = __builtin_shufflevector(alo, ahi, 0, 1, 2, 3, 4, 5, 6, 7,
                                     8, 9, 10, 11, 12, 13, 14, 15);
    __builtin_prefetch(arow + kt + 64 + halfk, 0, 1);   // global_prefetch_b8

    // B fragments (LDS): lane = N column, K in table order
    const _Float16* bl = smem + cur * BUFH + (lane & 15);
#pragma unroll
    for (int t = 0; t < 4; ++t) {
      v16h b;
#pragma unroll
      for (int s = 0; s < 8; ++s) {
        b[s]     = bl[(halfk + s) * BSTRIDE + t * 16];
        b[s + 8] = bl[(16 + halfk + s) * BSTRIDE + t * 16];
      }
      acc[t] = __builtin_amdgcn_wmma_f32_16x16x32_f16(
          /*neg_a=*/false, a, /*neg_b=*/false, b,
          /*c_mod=*/(short)0, acc[t], /*reuse_a=*/false, /*reuse_b=*/false);
    }
    __syncthreads();             // everyone done reading cur before overwrite
    cur ^= 1;
  }
}

// ---------------------------------------------------------------------------
// GEMM1: H1 = A1 @ W1; epilogue = exact GELU then emit augmented A2 row
// (silu(g) + 6 basis values) in f16.
// ---------------------------------------------------------------------------
__global__ __launch_bounds__(256) void kan_gemm1(const _Float16* __restrict__ A1,
                                                 const _Float16* __restrict__ W1,
                                                 _Float16* __restrict__ A2) {
  __shared__ _Float16 smem[2 * BUFH];
  const int wave  = threadIdx.x >> 5;
  const int mBase = (blockIdx.x >> 4) * 128 + wave * 16;  // 16 M blocks
  const int nBase = (blockIdx.x & 15) * 64;               // 16 N blocks

  v8f acc[4];
#pragma unroll
  for (int t = 0; t < 4; ++t)
#pragma unroll
    for (int e = 0; e < 8; ++e) acc[t][e] = 0.0f;

  kan_wmma_block<K1, HID>(A1, W1, smem, mBase, nBase, acc);

  // C/D layout: lanes 0-15 -> M = v, N = lane; lanes 16-31 -> M = 8+v, N = lane-16
  const int lane  = threadIdx.x & 31;
  const int col   = lane & 15;
  const int rhalf = (lane >> 4) * 8;
#pragma unroll
  for (int t = 0; t < 4; ++t) {
    const int o = nBase + t * 16 + col;      // layer-1 output feature
#pragma unroll
    for (int v = 0; v < 8; ++v) {
      const int row = mBase + rhalf + v;     // chunk-local token
      float g = kan_gelu(acc[t][v]);
      _Float16* dst = A2 + (size_t)row * K2 + (size_t)o * 7;
      dst[0] = (_Float16)kan_silu(g);
      float bas[6];
      kan_basis(g, bas);
#pragma unroll
      for (int j = 0; j < 6; ++j) dst[1 + j] = (_Float16)bas[j];
    }
  }
}

// ---------------------------------------------------------------------------
// GEMM2: Y = A2 @ W2 -> fp32 output.
// ---------------------------------------------------------------------------
__global__ __launch_bounds__(256) void kan_gemm2(const _Float16* __restrict__ A2,
                                                 const _Float16* __restrict__ W2,
                                                 float* __restrict__ out, int tok0) {
  __shared__ _Float16 smem[2 * BUFH];
  const int wave  = threadIdx.x >> 5;
  const int mBase = (blockIdx.x >> 2) * 128 + wave * 16;  // 16 M blocks
  const int nBase = (blockIdx.x & 3) * 64;                // 4 N blocks

  v8f acc[4];
#pragma unroll
  for (int t = 0; t < 4; ++t)
#pragma unroll
    for (int e = 0; e < 8; ++e) acc[t][e] = 0.0f;

  kan_wmma_block<K2, DIN>(A2, W2, smem, mBase, nBase, acc);

  const int lane  = threadIdx.x & 31;
  const int col   = lane & 15;
  const int rhalf = (lane >> 4) * 8;
#pragma unroll
  for (int t = 0; t < 4; ++t) {
    const int o = nBase + t * 16 + col;
#pragma unroll
    for (int v = 0; v < 8; ++v) {
      const int row = mBase + rhalf + v;
      out[(size_t)(tok0 + row) * DIN + o] = acc[t][v];
    }
  }
}

// ---------------------------------------------------------------------------
// Launcher. Workspace layout (f16, all 16B-aligned):
//   W1: K1*HID = 3.5 MB | W2: K2*DIN = 3.5 MB | A1: CHUNK*K1 = 7 MB
//   A2: CHUNK*K2 = 28 MB   -> total 42 MB.
// ---------------------------------------------------------------------------
extern "C" void kernel_launch(void* const* d_in, const int* in_sizes, int n_in,
                              void* d_out, int out_size, void* d_ws, size_t ws_size,
                              hipStream_t stream) {
  (void)in_sizes; (void)n_in; (void)out_size; (void)ws_size;
  const float* x   = (const float*)d_in[0];
  const float* bw1 = (const float*)d_in[1];
  const float* sw1 = (const float*)d_in[2];
  const float* sc1 = (const float*)d_in[3];
  const float* bw2 = (const float*)d_in[4];
  const float* sw2 = (const float*)d_in[5];
  const float* sc2 = (const float*)d_in[6];
  float* out = (float*)d_out;

  char* ws = (char*)d_ws;
  const size_t szW1 = (size_t)K1 * HID * sizeof(_Float16);   // 3,670,016
  const size_t szW2 = (size_t)K2 * DIN * sizeof(_Float16);   // 3,670,016
  const size_t szA1 = (size_t)CHUNK * K1 * sizeof(_Float16); // 7,340,032
  _Float16* W1 = (_Float16*)(ws);
  _Float16* W2 = (_Float16*)(ws + szW1);
  _Float16* A1 = (_Float16*)(ws + szW1 + szW2);
  _Float16* A2 = (_Float16*)(ws + szW1 + szW2 + szA1);

  kan_pack_w1<<<(HID * DIN) / 256, 256, 0, stream>>>(bw1, sw1, sc1, W1);
  kan_pack_w2<<<(DIN * HID) / 256, 256, 0, stream>>>(bw2, sw2, sc2, W2);

  for (int c = 0; c < NCHUNK; ++c) {
    const int tok0 = c * CHUNK;
    kan_build_a1<<<(CHUNK * DIN) / 256, 256, 0, stream>>>(x, A1, tok0);
    kan_gemm1<<<16 * 16, 256, 0, stream>>>(A1, W1, A2);   // 128x64 tiles
    kan_gemm2<<<16 * 4, 256, 0, stream>>>(A2, W2, out, tok0);
  }
}
